// GineBlock_12180527252066
// MI455X (gfx1250) — matrix operand order repaired and checked by reading, not compiled, not run on hardware
//
#include <hip/hip_runtime.h>
#include <hip/hip_bf16.h>
#include <stdint.h>

#define CDIM 256
#define NTILES 16   // 16 column tiles of 16
#define KBLKS 8     // 8 K-blocks of 32
#define NFRAG (KBLKS * NTILES)                // 128 B-fragments per stripe
#define WP_DWORDS (KBLKS * NTILES * 32 * 8)   // 32768 dwords = 128 KB

typedef __bf16 v16bf __attribute__((ext_vector_type(16)));
typedef float  v8f   __attribute__((ext_vector_type(8)));
typedef float  v4f   __attribute__((ext_vector_type(4)));

__device__ __forceinline__ uint32_t f2bf_u(float f) {
    uint32_t u = __float_as_uint(f);
    return (u + 0x7FFFu + ((u >> 16) & 1u)) >> 16;   // round-to-nearest-even
}

// ---------------------------------------------------------------------------
// 1) Edge scatter: z[dst] += relu(x[src] + edge_attr). z pre-initialized to x.
//    One thread per (edge, 128-channel chunk): E*2 threads.
//    edge_attr is streamed exactly once -> non-temporal loads so it does not
//    evict the L2-resident x (gather reuse) and z (atomic destination).
// ---------------------------------------------------------------------------
__global__ __launch_bounds__(256) void k_scatter(
    const float* __restrict__ x, const long long* __restrict__ ei,
    const float* __restrict__ ea, float* __restrict__ z, int E)
{
    int idx = blockIdx.x * blockDim.x + threadIdx.x;
    if (idx >= E * 2) return;
    int e  = idx >> 1;
    int cb = (idx & 1) * 128;
    size_t src = (size_t)ei[e];
    size_t dst = (size_t)ei[(size_t)E + e];
    const v4f* xs = (const v4f*)(x + src * CDIM + cb);
    const v4f* es = (const v4f*)(ea + (size_t)e * CDIM + cb);
    float* zd = z + dst * CDIM + cb;
    #pragma unroll 8
    for (int i = 0; i < 32; ++i) {
        v4f a = xs[i];
        v4f b = __builtin_nontemporal_load(es + i);
        float m0 = a.x + b.x; m0 = m0 > 0.f ? m0 : 0.f;
        float m1 = a.y + b.y; m1 = m1 > 0.f ? m1 : 0.f;
        float m2 = a.z + b.z; m2 = m2 > 0.f ? m2 : 0.f;
        float m3 = a.w + b.w; m3 = m3 > 0.f ? m3 : 0.f;
        atomicAdd(zd + i * 4 + 0, m0);
        atomicAdd(zd + i * 4 + 1, m1);
        atomicAdd(zd + i * 4 + 2, m2);
        atomicAdd(zd + i * 4 + 3, m3);
    }
}

// ---------------------------------------------------------------------------
// 2) Pack W (fp32 row-major [K=256][N=256]) into WMMA-B fragment order:
//    Wp[(f*32 + lane)*8 + v] = {bf16 W[k][n], bf16 W[k+1][n]},  f = kb*16+nt
//    with n = nt*16 + lane%16, k = kb*32 + (lane/16)*8 + 2v + (v/4)*8.
// ---------------------------------------------------------------------------
__global__ __launch_bounds__(256) void k_pack_w(
    const float* __restrict__ W, uint32_t* __restrict__ Wp)
{
    int idx = blockIdx.x * blockDim.x + threadIdx.x;   // 32768 total
    if (idx >= WP_DWORDS) return;
    int v    = idx & 7;
    int lane = (idx >> 3) & 31;
    int nt   = (idx >> 8) & 15;
    int kb   = idx >> 12;
    int n = nt * 16 + (lane & 15);
    int k = kb * 32 + (lane >> 4) * 8 + 2 * v + (v >> 2) * 8;
    uint32_t lo = f2bf_u(W[(size_t)k * CDIM + n]);
    uint32_t hi = f2bf_u(W[(size_t)(k + 1) * CDIM + n]);
    Wp[idx] = lo | (hi << 16);
}

// ---------------------------------------------------------------------------
// WMMA matmul core: one wave computes a full 16-row x 256-col stripe.
// B fragments stream from LDS through a 4-deep rotating register buffer so
// the DS pipe stays 3 fragments ahead of the WMMA pipe (no dscnt-0 drains).
// AF32=true : A read from fp32 row-major, converted to bf16 via native casts.
// AF32=false: A read from bf16 row-major (two b128 loads per fragment).
// ---------------------------------------------------------------------------
template <bool AF32>
__device__ __forceinline__ void matmul_stripe(
    const void* __restrict__ Abase, const uint32_t* WpL,
    int stripe, int lane, v8f (&acc)[NTILES])
{
    const int hf = lane >> 4;
    const int nn = lane & 15;

    union Bfrag { uint4 q[2]; v16bf v; };
    Bfrag bfr[4];

    auto loadB = [&](int f) {
        const uint32_t* p = WpL + ((unsigned)f * 32 + lane) * 8;
        bfr[f & 3].q[0] = *(const uint4*)(p);
        bfr[f & 3].q[1] = *(const uint4*)(p + 4);
    };
    auto loadA = [&](int kb) -> v16bf {
        if constexpr (AF32) {
            const float* arow = (const float*)Abase +
                (size_t)(stripe * 16 + nn) * CDIM + kb * 32 + hf * 8;
            v4f f0 = *(const v4f*)(arow);
            v4f f1 = *(const v4f*)(arow + 4);
            v4f f2 = *(const v4f*)(arow + 16);
            v4f f3 = *(const v4f*)(arow + 20);
            v16bf av;
            av[0]  = (__bf16)f0.x; av[1]  = (__bf16)f0.y;
            av[2]  = (__bf16)f0.z; av[3]  = (__bf16)f0.w;
            av[4]  = (__bf16)f1.x; av[5]  = (__bf16)f1.y;
            av[6]  = (__bf16)f1.z; av[7]  = (__bf16)f1.w;
            av[8]  = (__bf16)f2.x; av[9]  = (__bf16)f2.y;
            av[10] = (__bf16)f2.z; av[11] = (__bf16)f2.w;
            av[12] = (__bf16)f3.x; av[13] = (__bf16)f3.y;
            av[14] = (__bf16)f3.z; av[15] = (__bf16)f3.w;
            return av;
        } else {
            union { uint4 q[2]; v16bf v; } af;
            const unsigned short* arow = (const unsigned short*)Abase +
                (size_t)(stripe * 16 + nn) * CDIM + kb * 32 + hf * 8;
            af.q[0] = *(const uint4*)(arow);
            af.q[1] = *(const uint4*)(arow + 16);   // +16 elems = +32 B
            return af.v;
        }
    };

    loadB(0); loadB(1); loadB(2);   // prime the pipeline
    #pragma unroll
    for (int kb = 0; kb < KBLKS; ++kb) {
        v16bf av = loadA(kb);
        #pragma unroll
        for (int nt = 0; nt < NTILES; ++nt) {
            const int f = kb * NTILES + nt;
            if (f + 3 < NFRAG) loadB(f + 3);
            acc[nt] = __builtin_amdgcn_wmma_f32_16x16x32_bf16(
                false, av, false, bfr[f & 3].v, (short)0,
                kb == 0 ? (v8f)(0.f) : acc[nt], false, false);
        }
    }
}

// ---------------------------------------------------------------------------
// 3) GEMM1: h1 = relu(z @ W1 + b1), z fp32 in, output bf16 row-major.
//    Packed W1 staged once per block into 128 KB LDS.
// ---------------------------------------------------------------------------
__global__ __launch_bounds__(256) void k_gemm_relu(
    const float* __restrict__ z, const uint32_t* __restrict__ Wp,
    const float* __restrict__ bias, unsigned short* __restrict__ Ob, int nstripes)
{
    __shared__ uint32_t WpL[WP_DWORDS];   // 128 KB of the 320 KB WGP LDS
    {
        const uint4* s4 = (const uint4*)Wp;
        uint4* d4 = (uint4*)WpL;
        #pragma unroll
        for (int i = 0; i < WP_DWORDS / 4 / 256; ++i)
            d4[threadIdx.x + i * 256] = s4[threadIdx.x + i * 256];
    }
    __syncthreads();

    const int wave = threadIdx.x >> 5;
    const int lane = threadIdx.x & 31;
    const int stripe = blockIdx.x * 8 + wave;
    if (stripe >= nstripes) return;          // wave-uniform, after barrier
    const int hf = lane >> 4;
    const int nn = lane & 15;

    v8f acc[NTILES];
    matmul_stripe<true>(z, WpL, stripe, lane, acc);

    // D layout: lane holds col n=nn, row m = v + 8*hf
    #pragma unroll
    for (int nt = 0; nt < NTILES; ++nt) {
        float b = bias[nt * 16 + nn];
        #pragma unroll
        for (int v = 0; v < 8; ++v) {
            float val = acc[nt][v] + b;
            val = val > 0.f ? val : 0.f;
            size_t row = (size_t)(stripe * 16 + v + 8 * hf);
            Ob[row * CDIM + nt * 16 + nn] = (unsigned short)f2bf_u(val);
        }
    }
}

// ---------------------------------------------------------------------------
// 4) GEMM2 + fused residual + LayerNorm:
//    out = LN(x + relu(h1 @ W2 + b2)) * gamma + beta
// ---------------------------------------------------------------------------
__global__ __launch_bounds__(256) void k_gemm_ln(
    const unsigned short* __restrict__ h1, const uint32_t* __restrict__ Wp,
    const float* __restrict__ bias, const float* __restrict__ x,
    const float* __restrict__ gamma, const float* __restrict__ beta,
    float* __restrict__ out, int nstripes)
{
    __shared__ uint32_t WpL[WP_DWORDS];
    {
        const uint4* s4 = (const uint4*)Wp;
        uint4* d4 = (uint4*)WpL;
        #pragma unroll
        for (int i = 0; i < WP_DWORDS / 4 / 256; ++i)
            d4[threadIdx.x + i * 256] = s4[threadIdx.x + i * 256];
    }
    __syncthreads();

    const int wave = threadIdx.x >> 5;
    const int lane = threadIdx.x & 31;
    const int stripe = blockIdx.x * 8 + wave;
    if (stripe >= nstripes) return;
    const int hf = lane >> 4;
    const int nn = lane & 15;

    v8f acc[NTILES];
    matmul_stripe<false>(h1, WpL, stripe, lane, acc);

    // t = x + relu(h + b2), in place in acc
    #pragma unroll
    for (int nt = 0; nt < NTILES; ++nt) {
        float b = bias[nt * 16 + nn];
        #pragma unroll
        for (int v = 0; v < 8; ++v) {
            float h = acc[nt][v] + b;
            h = h > 0.f ? h : 0.f;
            size_t row = (size_t)(stripe * 16 + v + 8 * hf);
            acc[nt][v] = x[row * CDIM + nt * 16 + nn] + h;
        }
    }

    // Row stats: rows m = v + 8*hf live across the 16 lanes of each half.
    float mu[8], rs[8];
    #pragma unroll
    for (int v = 0; v < 8; ++v) {
        float s = 0.f, q = 0.f;
        #pragma unroll
        for (int nt = 0; nt < NTILES; ++nt) {
            float t = acc[nt][v];
            s += t; q += t * t;
        }
        #pragma unroll
        for (int m = 1; m < 16; m <<= 1) {
            s += __shfl_xor(s, m, 32);
            q += __shfl_xor(q, m, 32);
        }
        float mean = s * (1.f / 256.f);
        float var  = q * (1.f / 256.f) - mean * mean;
        mu[v] = mean;
        rs[v] = rsqrtf(var + 1e-5f);
    }

    #pragma unroll
    for (int nt = 0; nt < NTILES; ++nt) {
        float g  = gamma[nt * 16 + nn];
        float be = beta[nt * 16 + nn];
        #pragma unroll
        for (int v = 0; v < 8; ++v) {
            size_t row = (size_t)(stripe * 16 + v + 8 * hf);
            float o = (acc[nt][v] - mu[v]) * rs[v] * g + be;
            __builtin_nontemporal_store(o, out + row * CDIM + nt * 16 + nn);
        }
    }
}

// ---------------------------------------------------------------------------
extern "C" void kernel_launch(void* const* d_in, const int* in_sizes, int n_in,
                              void* d_out, int out_size, void* d_ws, size_t ws_size,
                              hipStream_t stream)
{
    const float*     x     = (const float*)d_in[0];
    const long long* ei    = (const long long*)d_in[1];
    const float*     ea    = (const float*)d_in[2];
    const float*     W1    = (const float*)d_in[3];
    const float*     b1    = (const float*)d_in[4];
    const float*     W2    = (const float*)d_in[5];
    const float*     b2    = (const float*)d_in[6];
    const float*     gamma = (const float*)d_in[7];
    const float*     beta  = (const float*)d_in[8];

    const int N = in_sizes[0] / CDIM;
    const int E = in_sizes[2] / CDIM;
    const int nstripes = (N + 15) / 16;     // N=50000 -> 3125 exact

    // Workspace layout
    char* ws = (char*)d_ws;
    size_t zN = (size_t)N * CDIM;
    float*          z   = (float*)ws;                               // zN*4 B
    unsigned short* h1  = (unsigned short*)(ws + zN * 4);           // zN*2 B
    uint32_t*       Wp1 = (uint32_t*)(ws + zN * 6);                 // 128 KB
    uint32_t*       Wp2 = Wp1 + WP_DWORDS;                          // 128 KB

    // z = x  (aggr accumulator initialized with residual source)
    hipMemcpyAsync(z, x, zN * sizeof(float), hipMemcpyDeviceToDevice, stream);

    // z += scatter-add of relu(x[src] + edge_attr)
    k_scatter<<<(E * 2 + 255) / 256, 256, 0, stream>>>(x, ei, ea, z, E);

    // packed bf16 weights
    k_pack_w<<<WP_DWORDS / 256, 256, 0, stream>>>(W1, Wp1);
    k_pack_w<<<WP_DWORDS / 256, 256, 0, stream>>>(W2, Wp2);

    // h1 = relu(z @ W1 + b1)   (fp32 A converted on the fly)
    k_gemm_relu<<<(nstripes + 7) / 8, 256, 0, stream>>>(z, Wp1, b1, h1, nstripes);

    // out = LN(x + relu(h1 @ W2 + b2))
    k_gemm_ln<<<(nstripes + 7) / 8, 256, 0, stream>>>(h1, Wp2, b2, x, gamma, beta,
                                                      (float*)d_out, nstripes);
}
